// TnoInvTime_15384572854670
// MI455X (gfx1250) — matrix operand order repaired and checked by reading, not compiled, not run on hardware
//
#include <hip/hip_runtime.h>

// ---------------------------------------------------------------------------
// TNO (Toeplitz neural operator): RPE-MLP builds kernel table a[h,2N,d],
// then out = Toeplitz(a) @ x as block-Toeplitz WMMA f16 matmuls with fp32
// accumulation (v_wmma_f32_16x16x32_f16, wave32 / gfx1250).
//
// v3: operand roles swapped (A = x slices, B = Toeplitz kernel). The kernel
// table is stored REVERSED (ar0[i] = a[(4095-i) mod 2N]) plus a half-shifted
// copy (ar1[i] = ar0[i+1]) so every lane's B fragment is one contiguous,
// ascending, >=4B-aligned 32-byte load. No LDS, no bpermute, no barriers in
// the conv inner loop -- just 3 vector loads + 1 WMMA per K=32 step.
// ---------------------------------------------------------------------------

#define BB 2
#define HH 8
#define NN 4096
#define DD 64
#define NCH 512              // H*D
#define TWO_N 8192
#define AEXT_W 8224          // 2N + 32 wrap-duplicate
#define XEXT_W 4352          // 112 left pad + 4096 + 144 right pad
#define XPAD 112

typedef _Float16 half_t;
typedef _Float16 v8h  __attribute__((ext_vector_type(8)));
typedef _Float16 v16h __attribute__((ext_vector_type(16)));
typedef float    v8f  __attribute__((ext_vector_type(8)));

// 32B vector load with only 4B alignment guarantee (backend may split it)
struct __attribute__((packed, aligned(4))) v16h_pack { v16h v; };

// ---------------------------------------------------------------------------
// Kernel 1: zero-fill x_ext (pads must be 0 so out-of-range conv taps vanish)
// ---------------------------------------------------------------------------
__global__ __launch_bounds__(256) void zero_kernel(unsigned long long* __restrict__ p, int n)
{
    int i = blockIdx.x * 256 + threadIdx.x;
    if (i < n) p[i] = 0ULL;
}

// ---------------------------------------------------------------------------
// Kernel 2: pack x (B,H,N,D) f32 -> x_ext[(b*H+h)*D+d][XPAD + s] f16
// LDS 64x64 tile transpose: coalesced reads (d fast) and writes (s fast).
// ---------------------------------------------------------------------------
__global__ __launch_bounds__(256) void pack_x_kernel(const float* __restrict__ x,
                                                     half_t* __restrict__ x_ext)
{
    const int bh = blockIdx.x >> 6;          // 0..15  (b*H + h)
    const int s0 = (blockIdx.x & 63) * 64;   // s tile base
    __shared__ half_t tile[64][66];          // padded to dodge bank conflicts

#pragma unroll
    for (int it = 0; it < 16; ++it) {
        int idx = it * 256 + threadIdx.x;
        int si = idx >> 6, d = idx & 63;
        tile[d][si] = (half_t)x[((size_t)bh * NN + (s0 + si)) * DD + d];
    }
    __syncthreads();
#pragma unroll
    for (int it = 0; it < 16; ++it) {
        int idx = it * 256 + threadIdx.x;
        int d = idx >> 6, si = idx & 63;
        x_ext[((size_t)bh * DD + d) * XEXT_W + (XPAD + s0 + si)] = tile[d][si];
    }
}

// ---------------------------------------------------------------------------
// Kernel 3: RPE MLP, fp32. One position per wave; lane L holds features
// (2L, 2L+1). Weights are wave-uniform -> constant-cache friendly loads.
// Position -> input mapping (matches reference concat layout):
//   p==0 or p==N : u = 1.0            (zero_k slots)
//   1..N-1       : u =  1/(p+1)       (pos_idx)
//   N+1..2N-1    : u = -1/(2N+1-p)    (neg_idx)
// Scatters each f16 value into the reversed tables:
//   ar0[ch][(4095-p) mod 2N] = v   (+ wrap duplicate at i+8192 if i < 32)
//   ar1[ch][(4094-p) mod 2N] = v   (ar1[i] == ar0[i+1])
// ---------------------------------------------------------------------------
__global__ __launch_bounds__(256) void rpe_mlp_kernel(
    const float* __restrict__ w_in,  const float* __restrict__ b_in,
    const float* __restrict__ w_hid, const float* __restrict__ b_hid,
    const float* __restrict__ w_out, const float* __restrict__ b_out,
    half_t* __restrict__ ar0, half_t* __restrict__ ar1)
{
    const int lane = threadIdx.x & 31;
    const int p = (blockIdx.x * 256 + threadIdx.x) >> 5;   // 0..8191

    float u;
    if (p == 0 || p == NN)      u = 1.0f;
    else if (p < NN)            u = 1.0f / (float)(p + 1);
    else                        u = -1.0f / (float)(2 * NN + 1 - p);

    const int f0 = 2 * lane, f1 = f0 + 1;
    float h0 = u * w_in[f0] + b_in[f0];
    float h1 = u * w_in[f1] + b_in[f1];
    float g0, g1;

    for (int layer = 0; layer < 3; ++layer) {
        float ss = h0 * h0 + h1 * h1;
#pragma unroll
        for (int msk = 16; msk >= 1; msk >>= 1) ss += __shfl_xor(ss, msk, 32);
        float r = rsqrtf(ss * (1.0f / 64.0f) + 1e-8f);
        g0 = fmaxf(h0 * r, 0.0f);
        g1 = fmaxf(h1 * r, 0.0f);

        const float* W = w_hid + layer * 64 * 64;
        float a0 = b_hid[layer * 64 + f0];
        float a1 = b_hid[layer * 64 + f1];
        for (int k2 = 0; k2 < 32; ++k2) {
            float ga = __shfl(g0, k2, 32);
            float gb = __shfl(g1, k2, 32);
            const float2* r0 = (const float2*)(W + (2 * k2) * 64);
            const float2* r1 = (const float2*)(W + (2 * k2 + 1) * 64);
            float2 v0 = r0[lane];
            float2 v1 = r1[lane];
            a0 = fmaf(ga, v0.x, a0); a1 = fmaf(ga, v0.y, a1);
            a0 = fmaf(gb, v1.x, a0); a1 = fmaf(gb, v1.y, a1);
        }
        h0 = a0; h1 = a1;
    }

    // final rms + relu
    {
        float ss = h0 * h0 + h1 * h1;
#pragma unroll
        for (int msk = 16; msk >= 1; msk >>= 1) ss += __shfl_xor(ss, msk, 32);
        float r = rsqrtf(ss * (1.0f / 64.0f) + 1e-8f);
        g0 = fmaxf(h0 * r, 0.0f);
        g1 = fmaxf(h1 * r, 0.0f);
    }

    // output layer: 512 channels; lane handles ch = c*32 + lane, c = 0..15
    float acc[16];
#pragma unroll
    for (int c = 0; c < 16; ++c) acc[c] = b_out[c * 32 + lane];
    for (int k2 = 0; k2 < 32; ++k2) {
        float ga = __shfl(g0, k2, 32);
        float gb = __shfl(g1, k2, 32);
        const float* r0 = w_out + (2 * k2) * NCH;
        const float* r1 = w_out + (2 * k2 + 1) * NCH;
#pragma unroll
        for (int c = 0; c < 16; ++c) {
            acc[c] = fmaf(ga, r0[c * 32 + lane], acc[c]);
            acc[c] = fmaf(gb, r1[c * 32 + lane], acc[c]);
        }
    }

    const int i0 = (4095 - p) & (TWO_N - 1);   // reversed position (ar0)
    const int i1 = (4094 - p) & (TWO_N - 1);   // reversed, half-shifted (ar1)
#pragma unroll
    for (int c = 0; c < 16; ++c) {
        int ch = c * 32 + lane;                // ch = h*64 + d
        half_t v = (half_t)acc[c];
        half_t* row0 = ar0 + (size_t)ch * AEXT_W;
        half_t* row1 = ar1 + (size_t)ch * AEXT_W;
        row0[i0] = v;
        if (i0 < (AEXT_W - TWO_N)) row0[i0 + TWO_N] = v;   // wrap duplicate
        row1[i1] = v;
        if (i1 < (AEXT_W - TWO_N)) row1[i1 + TWO_N] = v;
    }
}

// ---------------------------------------------------------------------------
// Kernel 4: block-Toeplitz conv via v_wmma_f32_16x16x32_f16.
// Wave task = (h, d, 128-wide t block). D = A x B with
//   A[m=(b,j), k] = x[b, s0 + 16j + k]        (two aligned b128 per lane)
//   B[k, n]       = a[T0 + n - s0 - k]        (one ascending 32B load per
//                                              lane from the reversed table;
//                                              lane parity picks ar0/ar1 so
//                                              every load is >=4B aligned)
//   D[(b,j), n] accumulates out[b, T0 + 16j + n] over 132 K=32 steps.
// No LDS / no cross-lane ops: pure loads + WMMA, fully pipelineable.
// ---------------------------------------------------------------------------
__global__ __launch_bounds__(256) void tno_conv_kernel(
    const half_t* __restrict__ ar0, const half_t* __restrict__ ar1,
    const half_t* __restrict__ x_ext, float* __restrict__ out)
{
    const int lane = threadIdx.x & 31;
    const int wave = threadIdx.x >> 5;
    const int dgrp = blockIdx.x & 7;
    const int tb   = (blockIdx.x >> 3) & 31;
    const int h    = blockIdx.x >> 8;
    const int d    = dgrp * 8 + wave;
    const int T0   = tb * 128;

    const int mrow  = lane & 15;                 // A row id  -> (b, j)
    const int bcol  = mrow & 1;
    const int j     = mrow >> 1;
    const int n     = lane & 15;                 // B col id  (t offset)
    const int hiH   = lane >> 4;
    const int koffA = hiH * 16;                  // A: k offset (contiguous, matches B)
    const int koffB = hiH * 16;                  // B: k offset (contiguous, matches A)

    const half_t* xrow = x_ext + (size_t)((bcol * HH + h) * DD + d) * XEXT_W;

    // G(step) = 4095 - T0 - n + s0 + koffB  (s0 = -112 + 32*step).
    // Parity of G == (n+1)&1: n even -> G odd -> use ar1 at G-1 (ar1[i]=ar0[i+1]).
    const size_t aoff = (size_t)(h * DD + d) * AEXT_W;
    const half_t* arow = ((n & 1) ? ar0 : ar1) + aoff;
    const int G0 = 4095 - T0 - n - 112 + koffB - ((n & 1) ? 0 : 1);

    v8f acc = {};
    for (int step = 0; step < 132; ++step) {
        const int xo = 32 * step + 16 * j + koffA;          // x_ext half index
        const v8h lo = *(const v8h*)(xrow + xo);            // k = koffA .. +7
        const v8h hi = *(const v8h*)(xrow + xo + 8);        // k = koffA+8 .. +15
        const v16h af = __builtin_shufflevector(lo, hi,
            0, 1, 2, 3, 4, 5, 6, 7, 8, 9, 10, 11, 12, 13, 14, 15);

        // bf[q] = a[(T0 + n - s0 - (q + koffB)) mod 2N], pre-reversed in memory
        const v16h bf = ((const v16h_pack*)(arow + (G0 + 32 * step)))->v;

        acc = __builtin_amdgcn_wmma_f32_16x16x32_f16(false, af, false, bf,
                                                     (short)0, acc, false, false);
    }

    // D layout: VGPR r -> row m = r + hiH*8 -> (b,j) = (m&1, m>>1); col = n
#pragma unroll
    for (int r = 0; r < 8; ++r) {
        const int mr = r + hiH * 8;
        const int br = mr & 1;
        const int jr = mr >> 1;
        out[((size_t)((br * HH + h) * NN) + T0 + 16 * jr + n) * DD + d] = acc[r];
    }
}

// ---------------------------------------------------------------------------
extern "C" void kernel_launch(void* const* d_in, const int* in_sizes, int n_in,
                              void* d_out, int out_size, void* d_ws, size_t ws_size,
                              hipStream_t stream)
{
    (void)in_sizes; (void)n_in; (void)out_size; (void)ws_size;
    const float* x     = (const float*)d_in[0];
    const float* w_in  = (const float*)d_in[1];
    const float* b_in  = (const float*)d_in[2];
    const float* w_hid = (const float*)d_in[3];
    const float* b_hid = (const float*)d_in[4];
    const float* w_out = (const float*)d_in[5];
    const float* b_out = (const float*)d_in[6];
    float* out = (float*)d_out;

    char* ws = (char*)d_ws;
    const size_t a_bytes = (size_t)NCH * AEXT_W * sizeof(half_t);   // 8,421,376
    half_t* ar0   = (half_t*)ws;
    half_t* ar1   = (half_t*)(ws + a_bytes);
    half_t* x_ext = (half_t*)(ws + 2 * a_bytes);                    // +8,912,896
    const size_t x_halves = (size_t)BB * HH * DD * XEXT_W;

    const int nz = (int)((x_halves * sizeof(half_t)) / 8);
    zero_kernel<<<(nz + 255) / 256, 256, 0, stream>>>((unsigned long long*)x_ext, nz);
    pack_x_kernel<<<BB * HH * (NN / 64), 256, 0, stream>>>(x, x_ext);
    rpe_mlp_kernel<<<(TWO_N * 32) / 256, 256, 0, stream>>>(w_in, b_in, w_hid, b_hid,
                                                           w_out, b_out, ar0, ar1);
    tno_conv_kernel<<<HH * 32 * 8, 256, 0, stream>>>(ar0, ar1, x_ext, out);
}
